// DigitByDigitValueHead_53566832116039
// MI455X (gfx1250) — matrix operand order repaired and checked by reading, not compiled, not run on hardware
//
#include <hip/hip_runtime.h>
#include <math.h>

#define D_MODEL 1024
#define DQ 256
#define DH 512
#define NPOS 6
#define BATCH 8
#define TSEQ 2048
#define BT (BATCH*TSEQ)

typedef __attribute__((ext_vector_type(16))) __bf16 v16bf;
typedef __attribute__((ext_vector_type(8)))  float  v8f;

static __device__ __forceinline__ __bf16 f2bf(float f){
  return (__bf16)f;   // RNE; lets backend use native bf16 cvt (packs to v_cvt_pk_bf16_f32)
}

static __device__ __forceinline__ float gelu_f(float x){
  return 0.5f * x * (1.0f + erff(x * 0.70710678118654752440f));
}

static __device__ __forceinline__ v16bf load_bvec(const __bf16* p){
  union { v16bf v; uint4 q[2]; } u;
  u.q[0] = *(const uint4*)(p);
  u.q[1] = *(const uint4*)(p + 8);
  return u.v;
}

// ---------------- prep kernels ----------------

__global__ void transpose_f32_to_bf16(const float* __restrict__ w, __bf16* __restrict__ wt,
                                      int K, int N){
  int idx = blockIdx.x * blockDim.x + threadIdx.x;
  if (idx >= K * N) return;
  int k = idx / N, n = idx % N;
  wt[(size_t)n * K + k] = f2bf(w[idx]);
}

__global__ void ctx_bias_kernel(const int* __restrict__ op_type,
                                const float* __restrict__ op_embed,
                                const float* __restrict__ pt_embed,
                                const float* __restrict__ ctx_w,
                                const float* __restrict__ ctx_b,
                                float* __restrict__ op_bias,
                                float* __restrict__ pt_bias){
  int n = blockIdx.x * blockDim.x + threadIdx.x;
  if (n >= D_MODEL) return;
  for (int b = 0; b < BATCH; ++b){
    int op = op_type[b];
    float s = 0.f;
    for (int q = 0; q < DQ; ++q)
      s += op_embed[op * DQ + q] * ctx_w[(size_t)(D_MODEL + q) * D_MODEL + n];
    op_bias[b * D_MODEL + n] = s;
  }
  for (int i = 0; i < 10; ++i){
    float s = ctx_b[n];
    for (int q = 0; q < DQ; ++q)
      s += pt_embed[i * DQ + q] * ctx_w[(size_t)(D_MODEL + DQ + q) * D_MODEL + n];
    pt_bias[i * D_MODEL + n] = s;
  }
}

__global__ void dig_pos_bias_kernel(const float* __restrict__ pos_embed,
                                    const float* __restrict__ dig_w1,
                                    const float* __restrict__ dig_b1,
                                    float* __restrict__ out){
  int idx = blockIdx.x * blockDim.x + threadIdx.x;
  if (idx >= NPOS * DH) return;
  int p = idx / DH, h = idx % DH;
  float s = dig_b1[idx];
  const float* wp = dig_w1 + ((size_t)p * (D_MODEL + DQ) + D_MODEL) * DH + h;
  for (int q = 0; q < DQ; ++q)
    s += pos_embed[p * DQ + q] * wp[(size_t)q * DH];
  out[idx] = s;
}

// ---- context: 32 rows/block, LDS-staged bf16 A, register LN + GELU -> bf16 ----

__global__ __launch_bounds__(512) void ctx_kernel(
    const float* __restrict__ hidden, const int* __restrict__ param_type,
    const __bf16* __restrict__ ctxWT,
    const float* __restrict__ op_bias, const float* __restrict__ pt_bias,
    const float* __restrict__ ctx_g, const float* __restrict__ ctx_beta,
    __bf16* __restrict__ ctx_out)
{
  __shared__ __bf16 sA[32][D_MODEL + 8];          // 66KB, b128-aligned rows (stride 2064B)
  __shared__ float sPartS[32][130], sPartQ[32][130];
  __shared__ float sSum[32][17], sSq[32][17];
  __shared__ float sMean[32], sRstd[32];
  __shared__ int   sPT[32];

  const int tid  = threadIdx.x;
  const int wave = tid >> 5, lane = tid & 31;
  const int lrow = lane & 15, kh = lane >> 4;
  const int row0 = blockIdx.x * 32;               // 32 tokens per block (T%32==0)
  const int bidx = row0 / TSEQ;
  const int waveM = (wave >> 3) * 16;             // 0 or 16
  const int waveN = wave & 7;
  const int c0   = waveN * 128;                   // each wave: 16 rows x 128 cols

  if (tid < 32) sPT[tid] = param_type[row0 + tid];

  // stage + convert A tile once (vs 8x redundant per-wave loads/converts)
  for (int idx = tid; idx < 32 * (D_MODEL / 2); idx += 512){
    int row = idx >> 9;
    int col = (idx & 511) * 2;
    float2 f = *(const float2*)(hidden + (size_t)(row0 + row) * D_MODEL + col);
    sA[row][col]     = f2bf(f.x);
    sA[row][col + 1] = f2bf(f.y);
  }
  __syncthreads();

  const v8f vzero = {0.f,0.f,0.f,0.f,0.f,0.f,0.f,0.f};
  v8f acc[8];
#pragma unroll
  for (int t = 0; t < 8; ++t) acc[t] = vzero;

  const __bf16* aRow = &sA[waveM + lrow][0];
  for (int k0 = 0; k0 < D_MODEL; k0 += 32){
    union { v16bf v; uint4 q[2]; } ua;
    ua.q[0] = *(const uint4*)(aRow + k0 + 16 * kh);       // ds_load_b128
    ua.q[1] = *(const uint4*)(aRow + k0 + 16 * kh + 8);
#pragma unroll
    for (int t = 0; t < 8; ++t){
      int n = c0 + t * 16 + lrow;
      v16bf b = load_bvec(ctxWT + (size_t)n * D_MODEL + k0 + 16 * kh);
      acc[t] = __builtin_amdgcn_wmma_f32_16x16x32_bf16(false, ua.v, false, b,
                                                       (short)0, acc[t], false, false);
    }
  }

  // bias into registers + per-lane LN partials
  float ps[8], pq[8];
#pragma unroll
  for (int r = 0; r < 8; ++r){ ps[r] = 0.f; pq[r] = 0.f; }
#pragma unroll
  for (int t = 0; t < 8; ++t){
    int n = c0 + t * 16 + lrow;
    float ob = op_bias[bidx * D_MODEL + n];
#pragma unroll
    for (int r = 0; r < 8; ++r){
      int m = waveM + r + 8 * kh;
      float v = acc[t][r] + ob + pt_bias[sPT[m] * D_MODEL + n];
      acc[t][r] = v;
      ps[r] += v; pq[r] += v * v;
    }
  }
  {
    int pcol = waveN * 16 + lrow;
#pragma unroll
    for (int r = 0; r < 8; ++r){
      int m = waveM + r + 8 * kh;
      sPartS[m][pcol] = ps[r];
      sPartQ[m][pcol] = pq[r];
    }
  }
  __syncthreads();

  {
    int row = tid >> 4, seg = tid & 15;
    float s = 0.f, ss = 0.f;
#pragma unroll
    for (int j = 0; j < 8; ++j){ s += sPartS[row][seg * 8 + j]; ss += sPartQ[row][seg * 8 + j]; }
    sSum[row][seg] = s; sSq[row][seg] = ss;
  }
  __syncthreads();
  if (tid < 32){
    float s = 0.f, ss = 0.f;
    for (int i = 0; i < 16; ++i){ s += sSum[tid][i]; ss += sSq[tid][i]; }
    float mean = s * (1.f / D_MODEL);
    float var  = ss * (1.f / D_MODEL) - mean * mean;
    sMean[tid] = mean;
    sRstd[tid] = rsqrtf(var + 1e-5f);
  }
  __syncthreads();

  // in-register LN + GELU, store bf16 context
#pragma unroll
  for (int t = 0; t < 8; ++t){
    int n = c0 + t * 16 + lrow;
    float gn = ctx_g[n], bn = ctx_beta[n];
#pragma unroll
    for (int r = 0; r < 8; ++r){
      int m = waveM + r + 8 * kh;
      float x = (acc[t][r] - sMean[m]) * sRstd[m] * gn + bn;
      ctx_out[(size_t)(row0 + m) * D_MODEL + n] = f2bf(gelu_f(x));
    }
  }
}

// ---- heads: 32 rows/block, GEMM + (LN)/GELU in-register + tiny matvec ----

__global__ __launch_bounds__(512) void heads_kernel(
    const __bf16* __restrict__ ctx, const __bf16* __restrict__ headWT,
    const float* __restrict__ sign_b1, const float* __restrict__ sign_g,
    const float* __restrict__ sign_beta, const float* __restrict__ sign_w2,
    const float* __restrict__ sign_b2,
    const float* __restrict__ digPB, const float* __restrict__ dig_g,
    const float* __restrict__ dig_beta, const float* __restrict__ dig_w2,
    const float* __restrict__ dig_b2,
    const float* __restrict__ aux_b1, const float* __restrict__ aux_w2,
    const float* __restrict__ aux_b2,
    float* __restrict__ out_sign, float* __restrict__ out_dig,
    float* __restrict__ out_aux)
{
  __shared__ float sH[32][DH + 4];                 // post-activation h (stage-2 input)
  __shared__ float sPartS[32][130], sPartQ[32][130];
  __shared__ float sSum[32][17], sSq[32][17];
  __shared__ float sMean[32], sRstd[32];

  const int tid  = threadIdx.x;
  const int wave = tid >> 5, lane = tid & 31;
  const int lrow = lane & 15, kh = lane >> 4;
  const int row0 = blockIdx.x * 32;
  const int head = blockIdx.y;                     // 0=sign, 1..6=digit p, 7=aux
  const int waveM = (wave >> 3) * 16;
  const int waveN = wave & 7;
  const int c0   = waveN * 64;                     // each wave: 16 rows x 64 cols

  const __bf16* WT = headWT + (size_t)head * DH * D_MODEL;
  const float* b1; const float* g = nullptr; const float* beta = nullptr;
  const float* w2; const float* b2; int no;
  if (head == 0){ b1 = sign_b1; g = sign_g; beta = sign_beta; w2 = sign_w2; b2 = sign_b2; no = 3; }
  else if (head <= 6){ int p = head - 1;
    b1 = digPB + p * DH; g = dig_g + p * DH; beta = dig_beta + p * DH;
    w2 = dig_w2 + (size_t)p * DH * 10; b2 = dig_b2 + p * 10; no = 10; }
  else { b1 = aux_b1; w2 = aux_w2; b2 = aux_b2; no = 1; }

  const v8f vzero = {0.f,0.f,0.f,0.f,0.f,0.f,0.f,0.f};
  v8f acc[4];
#pragma unroll
  for (int t = 0; t < 4; ++t) acc[t] = vzero;

  const __bf16* arow = ctx + (size_t)(row0 + waveM + lrow) * D_MODEL;
  for (int k0 = 0; k0 < D_MODEL; k0 += 32){
    union { v16bf v; uint4 q[2]; } ua;
    ua.q[0] = *(const uint4*)(arow + k0 + 16 * kh);
    ua.q[1] = *(const uint4*)(arow + k0 + 16 * kh + 8);
#pragma unroll
    for (int t = 0; t < 4; ++t){
      int n = c0 + t * 16 + lrow;
      v16bf b = load_bvec(WT + (size_t)n * D_MODEL + k0 + 16 * kh);
      acc[t] = __builtin_amdgcn_wmma_f32_16x16x32_bf16(false, ua.v, false, b,
                                                       (short)0, acc[t], false, false);
    }
  }

  float ps[8], pq[8];
#pragma unroll
  for (int r = 0; r < 8; ++r){ ps[r] = 0.f; pq[r] = 0.f; }
#pragma unroll
  for (int t = 0; t < 4; ++t){
    int n = c0 + t * 16 + lrow;
    float bb = b1[n];
#pragma unroll
    for (int r = 0; r < 8; ++r){
      float v = acc[t][r] + bb;
      acc[t][r] = v;
      ps[r] += v; pq[r] += v * v;
    }
  }

  if (head < 7){                                   // LayerNorm(DH) + GELU
    int pcol = waveN * 16 + lrow;
#pragma unroll
    for (int r = 0; r < 8; ++r){
      int m = waveM + r + 8 * kh;
      sPartS[m][pcol] = ps[r];
      sPartQ[m][pcol] = pq[r];
    }
    __syncthreads();
    {
      int row = tid >> 4, seg = tid & 15;
      float s = 0.f, ss = 0.f;
#pragma unroll
      for (int j = 0; j < 8; ++j){ s += sPartS[row][seg * 8 + j]; ss += sPartQ[row][seg * 8 + j]; }
      sSum[row][seg] = s; sSq[row][seg] = ss;
    }
    __syncthreads();
    if (tid < 32){
      float s = 0.f, ss = 0.f;
      for (int i = 0; i < 16; ++i){ s += sSum[tid][i]; ss += sSq[tid][i]; }
      float mean = s * (1.f / DH);
      float var  = ss * (1.f / DH) - mean * mean;
      sMean[tid] = mean; sRstd[tid] = rsqrtf(var + 1e-5f);
    }
    __syncthreads();
#pragma unroll
    for (int t = 0; t < 4; ++t){
      int n = c0 + t * 16 + lrow;
      float gn = g[n], bn = beta[n];
#pragma unroll
      for (int r = 0; r < 8; ++r){
        int m = waveM + r + 8 * kh;
        float x = (acc[t][r] - sMean[m]) * sRstd[m] * gn + bn;
        sH[m][n] = gelu_f(x);
      }
    }
  } else {                                         // aux head: GELU only
#pragma unroll
    for (int t = 0; t < 4; ++t){
      int n = c0 + t * 16 + lrow;
#pragma unroll
      for (int r = 0; r < 8; ++r){
        int m = waveM + r + 8 * kh;
        sH[m][n] = gelu_f(acc[t][r]);
      }
    }
  }
  __syncthreads();

  // tiny second matmul: [32,512] @ [512,no]
  const int elems = 32 * no;
  for (int e = tid; e < elems; e += 512){
    int row = e / no, o = e % no;
    float s = b2[o];
    for (int k = 0; k < DH; ++k)
      s += sH[row][k] * w2[k * no + o];
    int token = row0 + row;
    if (head == 0)       out_sign[token * 3 + o] = s;
    else if (head <= 6)  out_dig[((size_t)token * NPOS + (head - 1)) * 10 + o] = s;
    else                 out_aux[token] = s;
  }
}

// ---------------- launcher ----------------

extern "C" void kernel_launch(void* const* d_in, const int* in_sizes, int n_in,
                              void* d_out, int out_size, void* d_ws, size_t ws_size,
                              hipStream_t stream){
  const float* hidden    = (const float*)d_in[0];
  const int*   op_type   = (const int*)  d_in[1];
  const int*   param_type= (const int*)  d_in[2];
  const float* op_embed  = (const float*)d_in[3];
  const float* pt_embed  = (const float*)d_in[4];
  const float* pos_embed = (const float*)d_in[5];
  const float* ctx_w     = (const float*)d_in[6];
  const float* ctx_b     = (const float*)d_in[7];
  const float* ctx_g     = (const float*)d_in[8];
  const float* ctx_beta  = (const float*)d_in[9];
  const float* sign_w1   = (const float*)d_in[10];
  const float* sign_b1   = (const float*)d_in[11];
  const float* sign_g    = (const float*)d_in[12];
  const float* sign_beta = (const float*)d_in[13];
  const float* sign_w2   = (const float*)d_in[14];
  const float* sign_b2   = (const float*)d_in[15];
  const float* dig_w1    = (const float*)d_in[16];
  const float* dig_b1    = (const float*)d_in[17];
  const float* dig_g     = (const float*)d_in[18];
  const float* dig_beta  = (const float*)d_in[19];
  const float* dig_w2    = (const float*)d_in[20];
  const float* dig_b2    = (const float*)d_in[21];
  const float* aux_w1    = (const float*)d_in[22];
  const float* aux_b1    = (const float*)d_in[23];
  const float* aux_w2    = (const float*)d_in[24];
  const float* aux_b2    = (const float*)d_in[25];
  (void)in_sizes; (void)n_in; (void)out_size; (void)ws_size;

  float* out_sign = (float*)d_out;                      // [BT,3]
  float* out_dig  = out_sign + (size_t)BT * 3;          // [BT,6,10]
  float* out_aux  = out_dig + (size_t)BT * NPOS * 10;   // [BT]

  char* ws = (char*)d_ws;
  size_t off = 0;
  auto alloc = [&](size_t bytes){ size_t o = off; off = (off + bytes + 255) & ~(size_t)255; return o; };
  __bf16* ctxWT  = (__bf16*)(ws + alloc((size_t)D_MODEL * D_MODEL * 2));   // W^T [1024,1024]
  __bf16* headWT = (__bf16*)(ws + alloc((size_t)8 * DH * D_MODEL * 2));    // 8 heads W^T [512,1024]
  float*  op_bias= (float*) (ws + alloc((size_t)BATCH * D_MODEL * 4));
  float*  pt_bias= (float*) (ws + alloc((size_t)10 * D_MODEL * 4));
  float*  digPB  = (float*) (ws + alloc((size_t)NPOS * DH * 4));
  __bf16* ctxBF  = (__bf16*)(ws + alloc((size_t)BT * D_MODEL * 2));        // context bf16

  {
    int n = D_MODEL * D_MODEL;
    transpose_f32_to_bf16<<<(n + 255) / 256, 256, 0, stream>>>(ctx_w, ctxWT, D_MODEL, D_MODEL);
  }
  {
    int n = D_MODEL * DH;
    transpose_f32_to_bf16<<<(n + 255) / 256, 256, 0, stream>>>(
        sign_w1, headWT, D_MODEL, DH);
    for (int p = 0; p < NPOS; ++p)
      transpose_f32_to_bf16<<<(n + 255) / 256, 256, 0, stream>>>(
          dig_w1 + (size_t)p * (D_MODEL + DQ) * DH,
          headWT + (size_t)(1 + p) * DH * D_MODEL, D_MODEL, DH);
    transpose_f32_to_bf16<<<(n + 255) / 256, 256, 0, stream>>>(
        aux_w1, headWT + (size_t)7 * DH * D_MODEL, D_MODEL, DH);
  }
  ctx_bias_kernel<<<(D_MODEL + 255) / 256, 256, 0, stream>>>(
      op_type, op_embed, pt_embed, ctx_w, ctx_b, op_bias, pt_bias);
  dig_pos_bias_kernel<<<(NPOS * DH + 255) / 256, 256, 0, stream>>>(
      pos_embed, dig_w1, dig_b1, digPB);

  ctx_kernel<<<BT / 32, 512, 0, stream>>>(
      hidden, param_type, ctxWT, op_bias, pt_bias, ctx_g, ctx_beta, ctxBF);

  heads_kernel<<<dim3(BT / 32, 8), 512, 0, stream>>>(
      ctxBF, headWT,
      sign_b1, sign_g, sign_beta, sign_w2, sign_b2,
      digPB, dig_g, dig_beta, dig_w2, dig_b2,
      aux_b1, aux_w2, aux_b2,
      out_sign, out_dig, out_aux);
}
